// MultiHeadAttention_9474697855464
// MI455X (gfx1250) — matrix operand order repaired and checked by reading, not compiled, not run on hardware
//
#include <hip/hip_runtime.h>

// MI455X / gfx1250, wave32. All GEMMs via v_wmma_f32_16x16x32_bf16.
// B operands are pre-swizzled into fragment layout so every WMMA operand
// loads as global_load_b128 / ds_load_b128 pairs. The attention kernel
// double-buffers V fragments through LDS with GLOBAL_LOAD_ASYNC_TO_LDS_B128
// (ASYNCcnt pipeline, inline asm per CDNA5 ISA 15.18.3).
typedef __attribute__((ext_vector_type(16))) __bf16 v16bf;
typedef __attribute__((ext_vector_type(8)))  float  v8f;
typedef __attribute__((ext_vector_type(4)))  unsigned int v4u;

#define B_  4
#define E_  1024
#define S_  1024
#define H_  16
#define HS_ 64

__device__ __forceinline__ unsigned short f2bf(float f) {
  union { float f; unsigned u; } c; c.f = f;
  unsigned r = c.u + 0x7FFFu + ((c.u >> 16) & 1u);   // round-to-nearest-even
  return (unsigned short)(r >> 16);
}

union Frag {
  v16bf v;
  v4u q[2];
  unsigned short u[16];
};

__device__ __forceinline__ v8f zero8() {
  v8f z;
  for (int i = 0; i < 8; ++i) z[i] = 0.0f;
  return z;
}

__device__ __forceinline__ v8f wmma_bf16(v16bf a, v16bf b, v8f c) {
  // (neg_a, A, neg_b, B, c_mod, C, reuse_a, reuse_b)
  return __builtin_amdgcn_wmma_f32_16x16x32_bf16(false, a, false, b, (short)0, c,
                                                 false, false);
}

// Async-copy one 4KB V chunk (global, linear) into LDS, preserving layout.
// 8x GLOBAL_LOAD_ASYNC_TO_LDS_B128: lane L moves bytes [L*16, L*16+16) of each
// 512B slice.  Tracked by ASYNCcnt; consumer must s_wait_asynccnt.
__device__ __forceinline__ void async_copy_4k(unsigned ldsoff,
                                              const unsigned short* src,
                                              int lane) {
  unsigned long long ga =
      (unsigned long long)(src) + (unsigned)lane * 16u;
  unsigned lo = ldsoff + (unsigned)lane * 16u;
#pragma unroll
  for (int o = 0; o < 4096; o += 512) {
    asm volatile("global_load_async_to_lds_b128 %0, %1, off"
                 :: "v"(lo + (unsigned)o), "v"(ga + (unsigned long long)o)
                 : "memory");
  }
}

// ---------------------------------------------------------------------------
// Kernel 1: per-head projections q = Wq[h]@x, k = Wk[h]@x, v = Wv[h]@x.
//   Qt : (s, i) row-major            -> A-operand, 2x b128 per lane
//   Kf : B-frag swizzled [t16][i32-chunk][lane][e] for Q^T K
//   Vt : B-frag swizzled [t32-chunk][i16-tile][lane][e] for P * V
// ---------------------------------------------------------------------------
__global__ void __launch_bounds__(256) mha_proj(
    const float* __restrict__ Xq, const float* __restrict__ Xk,
    const float* __restrict__ Xv, const float* __restrict__ Wq,
    const float* __restrict__ Wk, const float* __restrict__ Wv,
    unsigned short* __restrict__ Qt, unsigned short* __restrict__ Kf,
    unsigned short* __restrict__ Vt)
{
  __shared__ float Wl[HS_][HS_ + 1];   // padded vs bank conflicts
  __shared__ float Xl[HS_][HS_];
  const int tid  = threadIdx.x;
  const int sIdx = tid & 63;           // t/s within 64-col chunk
  const int ig   = tid >> 6;           // 0..3 -> 16 output rows each
  const int bh   = blockIdx.x;         // b*H + h
  const int b    = bh >> 4, h = bh & 15;
  const int s0   = blockIdx.y * 64;
  const size_t xrow  = ((size_t)b * E_ + (size_t)h * HS_) * S_;
  const size_t wbase = (size_t)h * HS_ * HS_;
  const size_t hb    = (size_t)bh * S_ * HS_;

  for (int m = 0; m < 3; ++m) {
    const float* X = (m == 0) ? Xq : (m == 1) ? Xk : Xv;
    const float* W = (m == 0) ? Wq : (m == 1) ? Wk : Wv;
    for (int jj = 0; jj < 16; ++jj) {
      int j = ig * 16 + jj;
      Wl[j][sIdx] = W[wbase + (size_t)j * HS_ + sIdx];
      Xl[j][sIdx] = X[xrow + (size_t)j * S_ + s0 + sIdx];
    }
    __syncthreads();
    float acc[16];
    for (int r = 0; r < 16; ++r) acc[r] = 0.0f;
    for (int j = 0; j < HS_; ++j) {
      float xv = Xl[j][sIdx];
      for (int r = 0; r < 16; ++r)
        acc[r] += Wl[ig * 16 + r][j] * xv;
    }
    for (int r = 0; r < 16; ++r) {
      const int i = ig * 16 + r;
      const int t = s0 + sIdx;
      const unsigned short bf = f2bf(acc[r]);
      if (m == 0) {
        Qt[hb + (size_t)t * HS_ + i] = bf;
      } else if (m == 1) {
        // B[k=i][n=t]: lane = ((i>>3)&1)*16 + (t&15), e = (i&7) + ((i>>4)&1)*8
        const size_t a = (((size_t)(t >> 4) * 2 + (i >> 5)) * 32 +
                          (size_t)(((i >> 3) & 1) * 16 + (t & 15))) * 16 +
                         ((i & 7) + ((i >> 4) & 1) * 8);
        Kf[hb + a] = bf;
      } else {
        // B[k=t][n=i]: lane = ((t>>3)&1)*16 + (i&15), e = (t&7) + ((t>>4)&1)*8
        const size_t a = (((size_t)(t >> 5) * 4 + (i >> 4)) * 32 +
                          (size_t)(((t >> 3) & 1) * 16 + (i & 15))) * 16 +
                         ((t & 7) + ((t >> 4) & 1) * 8);
        Vt[hb + a] = bf;
      }
    }
    __syncthreads();
  }
}

// ---------------------------------------------------------------------------
// Kernel 2: convert Wo to bf16 once (row-major, A-operand friendly).
// ---------------------------------------------------------------------------
__global__ void __launch_bounds__(256) wo_convert(
    const float* __restrict__ Wo, unsigned short* __restrict__ WoB)
{
  int idx = blockIdx.x * 256 + threadIdx.x;
  WoB[idx] = f2bf(Wo[idx]);
}

// ---------------------------------------------------------------------------
// Kernel 3: inv[b,h,t] = 1 / sum_u exp(pre[t,u]).  One wave per 16 rows.
// ---------------------------------------------------------------------------
__global__ void __launch_bounds__(32) mha_denom(
    const unsigned short* __restrict__ Qt,
    const unsigned short* __restrict__ Kf, float* __restrict__ Inv)
{
  const int lane = threadIdx.x;
  const int bh   = blockIdx.y;
  const int s0   = blockIdx.x * 16;
  const size_t hb = (size_t)bh * S_ * HS_;
  const int n    = lane & 15;
  const int hi   = (lane >> 4) & 1;
  const int kloA = hi ? 8 : 0;     // A-operand K-half base (ISA 7.12.2)

  Frag aq0, aq1;                   // Q^T rows for this s-block (loop invariant)
  {
    const unsigned short* p = Qt + hb + (size_t)(s0 + n) * HS_;
    aq0.q[0] = *(const v4u*)(p + kloA);
    aq0.q[1] = *(const v4u*)(p + kloA + 16);
    aq1.q[0] = *(const v4u*)(p + 32 + kloA);
    aq1.q[1] = *(const v4u*)(p + 32 + kloA + 16);
  }
  float sums[8];
  for (int r = 0; r < 8; ++r) sums[r] = 0.0f;

  for (int t0 = 0; t0 < S_; t0 += 16) {
    const unsigned short* kb = Kf + hb + (size_t)(t0 >> 4) * 1024 + lane * 16;
    __builtin_prefetch(kb + 1024, 0, 1);               // next t-tile
    Frag bk0, bk1;
    bk0.q[0] = *(const v4u*)(kb);
    bk0.q[1] = *(const v4u*)(kb + 8);
    bk1.q[0] = *(const v4u*)(kb + 512);
    bk1.q[1] = *(const v4u*)(kb + 520);
    v8f c = zero8();
    c = wmma_bf16(aq0.v, bk0.v, c);
    c = wmma_bf16(aq1.v, bk1.v, c);
    for (int r = 0; r < 8; ++r)
      sums[r] += __expf(c[r] * 0.125f);                // 1/sqrt(64)
  }
  for (int r = 0; r < 8; ++r) {
    float v = sums[r];
    v += __shfl_xor(v, 1, 32);
    v += __shfl_xor(v, 2, 32);
    v += __shfl_xor(v, 4, 32);
    v += __shfl_xor(v, 8, 32);
    sums[r] = v;
  }
  if (n == 0) {
    for (int r = 0; r < 8; ++r)
      Inv[(size_t)bh * S_ + s0 + hi * 8 + r] = 1.0f / sums[r];
  }
}

// ---------------------------------------------------------------------------
// Kernel 4: O^T[s, i] = sum_t exp(pre[s,t]) * inv[t] * V[i,t].
// P routed through LDS (C/D -> A-operand).  V fragments double-buffered in
// LDS via GLOBAL_LOAD_ASYNC_TO_LDS_B128: chunk c's copy overlaps chunk c's
// QK^T WMMAs + exp pipeline; consumed with ds_load_b128 after
// s_wait_asynccnt 0.  Output written directly into the Wo-GEMM B-frag swizzle.
// ---------------------------------------------------------------------------
__global__ void __launch_bounds__(32) mha_attn(
    const unsigned short* __restrict__ Qt,
    const unsigned short* __restrict__ Kf,
    const unsigned short* __restrict__ Vt,
    const float* __restrict__ Inv, unsigned short* __restrict__ Obf)
{
  __shared__ __align__(16) unsigned short Pl[16 * 32];
  __shared__ __align__(16) unsigned short Vl[2][2048];   // 2 x 4KB V chunks
  const int lane = threadIdx.x;
  const int bh   = blockIdx.y;
  const int b    = bh >> 4, h = bh & 15;
  const int s0   = blockIdx.x * 16;
  const size_t hb = (size_t)bh * S_ * HS_;
  const int n    = lane & 15;
  const int hi   = (lane >> 4) & 1;
  const int kloA = hi ? 8 : 0;

  Frag aq0, aq1;
  {
    const unsigned short* p = Qt + hb + (size_t)(s0 + n) * HS_;
    aq0.q[0] = *(const v4u*)(p + kloA);
    aq0.q[1] = *(const v4u*)(p + kloA + 16);
    aq1.q[0] = *(const v4u*)(p + 32 + kloA);
    aq1.q[1] = *(const v4u*)(p + 32 + kloA + 16);
  }
  v8f accO[4];
  for (int j = 0; j < 4; ++j) accO[j] = zero8();
  const float* invp = Inv + (size_t)bh * S_;
  const unsigned short* vsrc = Vt + hb;
  const unsigned vlds0 = (unsigned)(unsigned long long)&Vl[0][0];
  const unsigned vlds1 = (unsigned)(unsigned long long)&Vl[1][0];

  // prologue: start async copy of V chunk 0
  async_copy_4k(vlds0, vsrc, lane);

  for (int t0 = 0; t0 < S_; t0 += 32) {
    const int buf = (t0 >> 5) & 1;
    // ---- P'' chunk: 16(s) x 32(t), exp*inv, bf16, into LDS ----
    // (independent of V; overlaps the in-flight async V copy)
    for (int half = 0; half < 2; ++half) {
      const int tt = t0 + half * 16;
      const unsigned short* kb = Kf + hb + (size_t)(tt >> 4) * 1024 + lane * 16;
      Frag bk0, bk1;
      bk0.q[0] = *(const v4u*)(kb);
      bk0.q[1] = *(const v4u*)(kb + 8);
      bk1.q[0] = *(const v4u*)(kb + 512);
      bk1.q[1] = *(const v4u*)(kb + 520);
      v8f c = zero8();
      c = wmma_bf16(aq0.v, bk0.v, c);
      c = wmma_bf16(aq1.v, bk1.v, c);
      const float iv = invp[tt + n];
      for (int r = 0; r < 8; ++r) {
        float p = __expf(c[r] * 0.125f) * iv;
        Pl[(r + hi * 8) * 32 + half * 16 + n] = f2bf(p);   // (s_local, t_local)
      }
    }
    __syncthreads();
    Frag ap;                                   // P as A-operand (16 x 32)
    {
      const unsigned short* pp = &Pl[n * 32];
      ap.q[0] = *(const v4u*)(pp + kloA);      // ds_load_b128
      ap.q[1] = *(const v4u*)(pp + kloA + 16);
    }
    // current V chunk is ready once ASYNCcnt drains; then kick off next chunk
    asm volatile("s_wait_asynccnt 0x0" ::: "memory");
    if (t0 + 32 < S_)
      async_copy_4k(buf ? vlds0 : vlds1, vsrc + ((t0 >> 5) + 1) * 2048, lane);
    const unsigned short* vl = &Vl[buf][0];
    for (int j = 0; j < 4; ++j) {
      Frag bv;                                 // V^T frag 32(t) x 16(i)
      bv.q[0] = *(const v4u*)(vl + j * 512 + lane * 16);
      bv.q[1] = *(const v4u*)(vl + j * 512 + lane * 16 + 8);
      accO[j] = wmma_bf16(ap.v, bv.v, accO[j]);
    }
    __syncthreads();
  }
  // store into B-frag swizzle of O: per b, [f_chunk(32)][s_tile(64)][lane][e]
  const int st = s0 >> 4;
  const size_t ob = (size_t)b * (size_t)(E_ * S_);
  for (int j = 0; j < 4; ++j) {
    for (int r = 0; r < 8; ++r) {
      const int f = h * HS_ + j * 16 + n;
      const int srow = r + hi * 8;
      const size_t a = (((size_t)(f >> 5) * 64 + st) * 32 +
                        (size_t)(((f >> 3) & 1) * 16 + srow)) * 16 + ((f & 7) + ((f >> 4) & 1) * 8);
      Obf[ob + a] = f2bf(accO[j][r]);
    }
  }
}

// ---------------------------------------------------------------------------
// Kernel 5: Out[b,e,s] = sum_f Wo[e,f] * O[b,f,s].  32 WMMAs per 16x16 tile;
// both operands load as b128 pairs.
// ---------------------------------------------------------------------------
__global__ void __launch_bounds__(32) mha_outproj(
    const unsigned short* __restrict__ WoB,
    const unsigned short* __restrict__ Obf, float* __restrict__ Out)
{
  const int lane = threadIdx.x;
  const int s0   = blockIdx.x * 16;
  const int e0   = blockIdx.y * 16;
  const int b    = blockIdx.z;
  const int n    = lane & 15;
  const int hi   = (lane >> 4) & 1;
  const int kloA = hi ? 8 : 0;
  const int st   = s0 >> 4;
  const size_t ob = (size_t)b * (size_t)(E_ * S_);

  v8f acc = zero8();
  for (int f0 = 0; f0 < E_; f0 += 32) {
    Frag a, bfrag;
    const unsigned short* wp = WoB + (size_t)(e0 + n) * E_ + f0;
    a.q[0] = *(const v4u*)(wp + kloA);
    a.q[1] = *(const v4u*)(wp + kloA + 16);
    const unsigned short* op =
        Obf + ob + (((size_t)(f0 >> 5) * 64 + st) * 32 + lane) * 16;
    __builtin_prefetch(op + 64 * 32 * 16, 0, 1);       // next f-chunk
    bfrag.q[0] = *(const v4u*)(op);
    bfrag.q[1] = *(const v4u*)(op + 8);
    acc = wmma_bf16(a.v, bfrag.v, acc);
  }
  for (int r = 0; r < 8; ++r)
    Out[ob + (size_t)(e0 + r + hi * 8) * S_ + s0 + n] = acc[r];
}

// ---------------------------------------------------------------------------
extern "C" void kernel_launch(void* const* d_in, const int* in_sizes, int n_in,
                              void* d_out, int out_size, void* d_ws,
                              size_t ws_size, hipStream_t stream)
{
  (void)in_sizes; (void)n_in; (void)out_size; (void)ws_size;
  const float* Xq = (const float*)d_in[0];
  const float* Xk = (const float*)d_in[1];
  const float* Xv = (const float*)d_in[2];
  const float* Wq = (const float*)d_in[3];
  const float* Wk = (const float*)d_in[4];
  const float* Wv = (const float*)d_in[5];
  const float* Wo = (const float*)d_in[6];
  float* Out = (float*)d_out;

  // Workspace carve-up (bf16 arrays are 8MB each; total ~34.25MB)
  char* ws = (char*)d_ws;
  const size_t SZ = (size_t)B_ * H_ * S_ * HS_;        // 4M elements
  unsigned short* Qt  = (unsigned short*)(ws);
  unsigned short* Kf  = (unsigned short*)(ws + SZ * 2);
  unsigned short* Vt  = (unsigned short*)(ws + SZ * 4);
  unsigned short* Obf = (unsigned short*)(ws + SZ * 6);
  unsigned short* WoB = (unsigned short*)(ws + SZ * 8);
  float* Inv = (float*)(ws + SZ * 8 + (size_t)E_ * E_ * 2);

  dim3 g1(B_ * H_, S_ / 64);
  mha_proj<<<g1, 256, 0, stream>>>(Xq, Xk, Xv, Wq, Wk, Wv, Qt, Kf, Vt);
  wo_convert<<<(E_ * E_) / 256, 256, 0, stream>>>(Wo, WoB);
  dim3 g2(S_ / 16, B_ * H_);
  mha_denom<<<g2, 32, 0, stream>>>(Qt, Kf, Inv);
  mha_attn<<<g2, 32, 0, stream>>>(Qt, Kf, Vt, Inv, Obf);
  dim3 g5(S_ / 16, E_ / 16, B_);
  mha_outproj<<<g5, 32, 0, stream>>>(WoB, Obf, Out);
}
